// HeteroRRT_76673756168223
// MI455X (gfx1250) — matrix operand order repaired and checked by reading, not compiled, not run on hardware
//
#include <hip/hip_runtime.h>
#include <cstdint>
#include <cstddef>

#define HH 4
#define DD 32
#define CC 128
#define SQD 0.17677669529663687f  // 1/sqrt(32)

#define AS1 __attribute__((address_space(1)))
#define AS3 __attribute__((address_space(3)))

#if defined(__has_builtin)
#if __has_builtin(__builtin_amdgcn_global_load_async_to_lds_b128)
#define HAVE_ASYNC_LDS 1
#endif
#endif

typedef __attribute__((ext_vector_type(16))) _Float16 v16h;
typedef __attribute__((ext_vector_type(8)))  float    v8f;
typedef int v4i_vs __attribute__((vector_size(16)));   // matches builtin param type

// ---- order-preserving float<->uint encode for atomic segment-max ----
static __device__ __forceinline__ unsigned int fenc(float f) {
  unsigned int u = __float_as_uint(f);
  return (u & 0x80000000u) ? ~u : (u | 0x80000000u);
}
static __device__ __forceinline__ float fdec(unsigned int u) {
  return (u & 0x80000000u) ? __uint_as_float(u & 0x7FFFFFFFu)
                           : __uint_as_float(~u);
}

// ---------------- fills ----------------
__global__ void k_fill_f32(float* p, float v, size_t n) {
  size_t i = (size_t)blockIdx.x * blockDim.x + threadIdx.x;
  if (i < n) p[i] = v;
}
__global__ void k_fill_u32(unsigned int* p, unsigned int v, size_t n) {
  size_t i = (size_t)blockIdx.x * blockDim.x + threadIdx.x;
  if (i < n) p[i] = v;
}

// ---------------- embedding + layernorm (one node per 128-thread block) ----------------
__global__ void k_embed_ln(const float* __restrict__ x, int fin,
                           const float* __restrict__ We, const float* __restrict__ be,
                           const float* __restrict__ g,  const float* __restrict__ bt,
                           float* __restrict__ out) {
  const int node = blockIdx.x, c = threadIdx.x;
  __shared__ float red[CC];
  float v = be[c];
  for (int k = 0; k < fin; ++k) v = fmaf(x[(size_t)node * fin + k], We[k * CC + c], v);
  red[c] = v; __syncthreads();
  for (int off = 64; off > 0; off >>= 1) { if (c < off) red[c] += red[c + off]; __syncthreads(); }
  float mean = red[0] * (1.0f / CC);
  __syncthreads();
  float d = v - mean;
  red[c] = d * d; __syncthreads();
  for (int off = 64; off > 0; off >>= 1) { if (c < off) red[c] += red[c + off]; __syncthreads(); }
  float var = red[0] * (1.0f / CC);
  out[(size_t)node * CC + c] = d * rsqrtf(var + 1e-5f) * g[c] + bt[c];
}

// ---------------- WMMA GEMM: Y[M,ncols] = act(X[M,128] @ W[128,ncols] + bias) ----------------
// Weight panel (128 x ncols, f32) is staged into LDS once per block via the CDNA5
// async global->LDS data mover (ASYNCcnt) when available. Each wave computes a
// 16-row x 32-col strip (two 16x16 WMMA tiles, A fragment reused across both).
// blockDim = (32, ncols/32); grid.x = M/16; dynamic LDS = 128*ncols*4 bytes.
__global__ void k_gemm(const float* __restrict__ X, const float* __restrict__ W,
                       const float* __restrict__ bias, float* __restrict__ Y,
                       int M, int ncols, int act) {
  extern __shared__ float wlds[];  // [128][ncols]
  const int nthr = blockDim.x * blockDim.y;
  const int tid  = threadIdx.y * 32 + threadIdx.x;

  // ---- stage W into LDS ----
#ifdef HAVE_ASYNC_LDS
  {
    const AS1 float* gw = (const AS1 float*)W;
    AS3 float*       lw = (AS3 float*)wlds;
    const int nvec = ncols * 32;  // 16-byte chunks in the 128 x ncols panel
    for (int c = tid; c < nvec; c += nthr)
      __builtin_amdgcn_global_load_async_to_lds_b128((AS1 v4i_vs*)(gw + (size_t)c * 4),
                                                     (AS3 v4i_vs*)(lw + (size_t)c * 4), 0, 0);
#if __has_builtin(__builtin_amdgcn_s_wait_asynccnt)
    __builtin_amdgcn_s_wait_asynccnt(0);
#else
    asm volatile("s_wait_asynccnt 0" ::: "memory");
#endif
  }
#else
  for (int c = tid; c < ncols * CC; c += nthr) wlds[c] = W[c];
#endif
  __syncthreads();

  const int lane = threadIdx.x;      // 0..31
  const int m    = lane & 15;
  const int hi   = lane >> 4;        // 0/1
  const int row0 = blockIdx.x * 16;
  const int colb = threadIdx.y * 32; // two 16-wide tiles: colb, colb+16
  int ra = row0 + m; if (ra > M - 1) ra = M - 1;
  const float* Xrow = X + (size_t)ra * CC;
  // speculative prefetch of the next 16-row A panel (global_prefetch_b8)
  __builtin_prefetch(Xrow + (size_t)16 * CC, 0, 1);

  v8f acc0 = {}, acc1 = {};
#pragma unroll
  for (int k0 = 0; k0 < CC; k0 += 32) {
    v16h a, b0, b1;
#pragma unroll
    for (int j = 0; j < 8; ++j) {
      // A 16x32 f16 layout: VGPR j holds K = (j>=4?16:0) + (hi?8:0) + 2*(j&3) + {0,1}
      const int ka = k0 + ((j & 4) << 2) + (hi << 3) + ((j & 3) << 1);
      a[2 * j]     = (_Float16)Xrow[ka];
      a[2 * j + 1] = (_Float16)Xrow[ka + 1];
      // B 32x16 f16 layout: VGPR j holds K = (j>=4?16:0) + (hi?8:0) + 2*(j&3) + {0,1}, N = lane&15
      const int kb = k0 + ((j & 4) << 2) + (hi << 3) + ((j & 3) << 1);
      b0[2 * j]     = (_Float16)wlds[kb * ncols + colb + m];
      b0[2 * j + 1] = (_Float16)wlds[(kb + 1) * ncols + colb + m];
      b1[2 * j]     = (_Float16)wlds[kb * ncols + colb + 16 + m];
      b1[2 * j + 1] = (_Float16)wlds[(kb + 1) * ncols + colb + 16 + m];
    }
    acc0 = __builtin_amdgcn_wmma_f32_16x16x32_f16(false, a, false, b0, (short)0, acc0, false, false);
    acc1 = __builtin_amdgcn_wmma_f32_16x16x32_f16(false, a, false, b1, (short)0, acc1, false, false);
  }
  const float bv0 = bias[colb + m];
  const float bv1 = bias[colb + 16 + m];
#pragma unroll
  for (int r = 0; r < 8; ++r) {
    const int row = row0 + r + 8 * hi;   // C/D layout: VGPR r -> M = r + 8*hi, N = lane&15
    if (row < M) {
      float y0 = acc0[r] + bv0;
      float y1 = acc1[r] + bv1;
      if (act == 1) { y0 = fmaxf(y0, 0.0f); y1 = fmaxf(y1, 0.0f); }
      Y[(size_t)row * ncols + colb + m]      = y0;
      Y[(size_t)row * ncols + colb + 16 + m] = y1;
    }
  }
}

// ---------------- exact GELU in-place ----------------
__global__ void k_gelu(float* p, size_t n) {
  size_t i = (size_t)blockIdx.x * blockDim.x + threadIdx.x;
  if (i < n) {
    float x = p[i];
    p[i] = 0.5f * x * (1.0f + erff(x * 0.7071067811865475f));
  }
}

// ---------------- edge pass 1: attention logits + segment max ----------------
// one wave per (edge, head); lane = feature f
__global__ void k_edge_al(const int* __restrict__ esrc, const int* __restrict__ edst, int E,
                          const float* __restrict__ K, const float* __restrict__ Q,
                          const float* __restrict__ Ar, const float* __restrict__ Pr,
                          float* __restrict__ albuf, unsigned int* __restrict__ segmax) {
  const int gt = blockIdx.x * blockDim.x + threadIdx.x;
  const int wid = gt >> 5, lane = gt & 31;
  if (wid >= E * HH) return;
  const int e = wid >> 2, h = wid & 3;
  const int src = esrc[e], dst = edst[e];
  const float* kp = K + (size_t)src * CC + h * DD;
  const float* qp = Q + (size_t)dst * CC + h * DD;
  const float* ar = Ar + (size_t)h * DD * DD;
  float kef = 0.0f;
#pragma unroll 8
  for (int d = 0; d < DD; ++d) kef = fmaf(kp[d], ar[d * DD + lane], kef);
  float s = qp[lane] * kef;
#pragma unroll
  for (int o = 16; o > 0; o >>= 1) s += __shfl_xor(s, o, 32);
  if (lane == 0) {
    const float al = Pr[h] * SQD * s;
    albuf[(size_t)e * HH + h] = al;
    atomicMax(segmax + (size_t)dst * HH + h, fenc(al));
  }
}

// ---------------- edge pass 2: exp(a - max) + segment sum ----------------
__global__ void k_edge_ex(const int* __restrict__ edst, int E,
                          float* __restrict__ albuf,
                          const unsigned int* __restrict__ segmax,
                          float* __restrict__ segsum) {
  const int i = blockIdx.x * blockDim.x + threadIdx.x;
  if (i >= E * HH) return;
  const int e = i >> 2, h = i & 3;
  const int dst = edst[e];
  const float mx = fdec(segmax[(size_t)dst * HH + h]);
  const float ex = expf(albuf[i] - mx);
  albuf[i] = ex;
  atomicAdd(segsum + (size_t)dst * HH + h, ex);
}

// ---------------- edge pass 3: message transform + weighted scatter ----------------
__global__ void k_edge_agg(const int* __restrict__ esrc, const int* __restrict__ edst, int E,
                           const float* __restrict__ V, const float* __restrict__ Mr,
                           const float* __restrict__ albuf, const float* __restrict__ segsum,
                           float* __restrict__ agg) {
  const int gt = blockIdx.x * blockDim.x + threadIdx.x;
  const int wid = gt >> 5, lane = gt & 31;
  if (wid >= E * HH) return;
  const int e = wid >> 2, h = wid & 3;
  const int src = esrc[e], dst = edst[e];
  const float w = albuf[(size_t)e * HH + h] / (segsum[(size_t)dst * HH + h] + 1e-16f);
  const float* vp = V + (size_t)src * CC + h * DD;
  const float* mr = Mr + (size_t)h * DD * DD;
  float vef = 0.0f;
#pragma unroll 8
  for (int d = 0; d < DD; ++d) vef = fmaf(vp[d], mr[d * DD + lane], vef);
  atomicAdd(agg + (size_t)dst * CC + h * DD + lane, w * vef);
}

// ---------------- skip-gate combine + leaky relu ----------------
// mode 0: x = lrelu(b*o + (1-b)*x); mode 1: += res; mode 2: += x (identity residual)
__global__ void k_combine(const float* __restrict__ o, float* __restrict__ x,
                          const float* __restrict__ res, const float* __restrict__ skipp,
                          int mode, size_t n) {
  size_t i = (size_t)blockIdx.x * blockDim.x + threadIdx.x;
  if (i >= n) return;
  const float beta = 1.0f / (1.0f + expf(-skipp[0]));
  const float xv = x[i];
  float v = beta * o[i] + (1.0f - beta) * xv;
  if (mode == 1) v += res[i];
  else if (mode == 2) v += xv;
  x[i] = (v >= 0.0f) ? v : 0.2f * v;
}

// ---------------- final head: out[n] = h[n,:64] . Wo2 + bo2 ----------------
__global__ void k_final(const float* __restrict__ h, const float* __restrict__ w2,
                        const float* __restrict__ b2, float* __restrict__ out, int n) {
  const int i = blockIdx.x * blockDim.x + threadIdx.x;
  if (i >= n) return;
  float s = b2[0];
#pragma unroll
  for (int j = 0; j < 64; ++j) s = fmaf(h[(size_t)i * 64 + j], w2[j], s);
  out[i] = s;
}

static inline unsigned cdiv(size_t a, unsigned b) { return (unsigned)((a + b - 1) / b); }

extern "C" void kernel_launch(void* const* d_in, const int* in_sizes, int n_in,
                              void* d_out, int out_size, void* d_ws, size_t ws_size,
                              hipStream_t stream) {
  (void)n_in; (void)out_size; (void)ws_size;
  // ---- inputs ----
  const float* x_in[3] = {(const float*)d_in[0], (const float*)d_in[1], (const float*)d_in[2]};
  const int fin[3] = {3, 3, 2};
  const float* Wem[3] = {(const float*)d_in[4],  (const float*)d_in[8],  (const float*)d_in[12]};
  const float* bem[3] = {(const float*)d_in[5],  (const float*)d_in[9],  (const float*)d_in[13]};
  const float* gln[3] = {(const float*)d_in[6],  (const float*)d_in[10], (const float*)d_in[14]};
  const float* bln[3] = {(const float*)d_in[7],  (const float*)d_in[11], (const float*)d_in[15]};
  const float* Wk = (const float*)d_in[20]; const float* bk = (const float*)d_in[21];
  const float* Wq = (const float*)d_in[22]; const float* bq = (const float*)d_in[23];
  const float* Wv = (const float*)d_in[24]; const float* bv = (const float*)d_in[25];
  const float* Wa = (const float*)d_in[26]; const float* ba = (const float*)d_in[27];
  const float* skip = (const float*)d_in[28];
  const float* Arel = (const float*)d_in[29]; const float* Mrel = (const float*)d_in[30];
  const float* Prel = (const float*)d_in[31];
  const float* Wres_s = (const float*)d_in[32]; const float* bres_s = (const float*)d_in[33];
  const float* Wres_o = (const float*)d_in[34]; const float* bres_o = (const float*)d_in[35];
  const float* Wo1 = (const float*)d_in[36]; const float* bo1 = (const float*)d_in[37];
  const float* Wo2 = (const float*)d_in[38]; const float* bo2 = (const float*)d_in[39];
  const int* eidx[8]; int Ecnt[8];
  for (int e = 0; e < 8; ++e) { eidx[e] = (const int*)d_in[40 + e]; Ecnt[e] = in_sizes[40 + e] / 2; }
  static const int ST[8] = {0, 0, 1, 2, 0, 1, 0, 0};  // edge src type
  static const int DT[8] = {0, 1, 0, 0, 0, 0, 1, 2};  // edge dst type
  // Node type 3 (Goal) never appears in any edge and never reaches the output -> skipped.
  int Nt[3] = {in_sizes[0] / 3, in_sizes[1] / 3, in_sizes[2] / 2};
  size_t nodeOff[3] = {0, (size_t)Nt[0], (size_t)Nt[0] + (size_t)Nt[1]};
  const size_t NTOT = nodeOff[2] + (size_t)Nt[2];
  size_t EALL = 0; for (int e = 0; e < 8; ++e) EALL += (size_t)Ecnt[e];

  // ---- workspace carve ----
  char* p = (char*)d_ws;
  auto carve = [&](size_t bytes) -> void* { void* r = (void*)p; p += (bytes + 255) & ~(size_t)255; return r; };
  float* xcur = (float*)carve(NTOT * CC * sizeof(float));
  float* xnew = (float*)carve(NTOT * CC * sizeof(float));
  float* Kb   = (float*)carve(NTOT * CC * sizeof(float));
  float* Qb   = (float*)carve(NTOT * CC * sizeof(float));
  float* Vb   = (float*)carve(NTOT * CC * sizeof(float));
  float* agg  = (float*)carve(NTOT * CC * sizeof(float));  // reused for res buffers + final hidden
  unsigned int* segmax = (unsigned int*)carve(NTOT * HH * sizeof(unsigned int));
  float* segsum = (float*)carve(NTOT * HH * sizeof(float));
  float* albuf  = (float*)carve(EALL * HH * sizeof(float));

  auto launch_gemm = [&](const float* Xp, const float* Wp, const float* bp, float* Yp,
                         int Mm, int nc, int act) {
    dim3 g(Mm / 16), b(32, nc / 32);
    size_t sh = (size_t)CC * nc * sizeof(float);
    k_gemm<<<g, b, sh, stream>>>(Xp, Wp, bp, Yp, Mm, nc, act);
  };

  // ---- embed + layernorm ----
  for (int t = 0; t < 3; ++t)
    k_embed_ln<<<Nt[t], CC, 0, stream>>>(x_in[t], fin[t], Wem[t], bem[t], gln[t], bln[t],
                                         xcur + nodeOff[t] * CC);

  // ---- 3 HGT layers ----
  for (int l = 0; l < 3; ++l) {
    // K/Q/V projections (WMMA)
    for (int t = 0; t < 3; ++t) {
      const size_t wo = ((size_t)l * 4 + t) * CC * CC, bo = ((size_t)l * 4 + t) * CC;
      const float* xin = xcur + nodeOff[t] * CC;
      launch_gemm(xin, Wk + wo, bk + bo, Kb + nodeOff[t] * CC, Nt[t], CC, 0);
      launch_gemm(xin, Wq + wo, bq + bo, Qb + nodeOff[t] * CC, Nt[t], CC, 0);
      launch_gemm(xin, Wv + wo, bv + bo, Vb + nodeOff[t] * CC, Nt[t], CC, 0);
    }
    // segment buffers
    const size_t nseg = NTOT * HH;
    k_fill_u32<<<cdiv(nseg, 256), 256, 0, stream>>>(segmax, 0x007FFFFFu /* enc(-inf) */, nseg);
    k_fill_f32<<<cdiv(nseg, 256), 256, 0, stream>>>(segsum, 0.0f, nseg);
    k_fill_f32<<<cdiv(NTOT * CC, 256), 256, 0, stream>>>(agg, 0.0f, NTOT * CC);
    // pass 1: logits + segment max
    size_t alOff = 0;
    for (int e = 0; e < 8; ++e) {
      const int E = Ecnt[e], st = ST[e], dt = DT[e];
      const size_t nw = (size_t)E * HH;
      k_edge_al<<<cdiv(nw * 32, 256), 256, 0, stream>>>(
          eidx[e], eidx[e] + E, E, Kb + nodeOff[st] * CC, Qb + nodeOff[dt] * CC,
          Arel + ((size_t)l * 8 + e) * HH * DD * DD, Prel + ((size_t)l * 8 + e) * HH,
          albuf + alOff, segmax + nodeOff[dt] * HH);
      alOff += nw;
    }
    // pass 2: exp + segment sum
    alOff = 0;
    for (int e = 0; e < 8; ++e) {
      const int E = Ecnt[e], dt = DT[e];
      const size_t nw = (size_t)E * HH;
      k_edge_ex<<<cdiv(nw, 256), 256, 0, stream>>>(
          eidx[e] + E, E, albuf + alOff, segmax + nodeOff[dt] * HH, segsum + nodeOff[dt] * HH);
      alOff += nw;
    }
    // pass 3: message transform + weighted scatter
    alOff = 0;
    for (int e = 0; e < 8; ++e) {
      const int E = Ecnt[e], st = ST[e], dt = DT[e];
      const size_t nw = (size_t)E * HH;
      k_edge_agg<<<cdiv(nw * 32, 256), 256, 0, stream>>>(
          eidx[e], eidx[e] + E, E, Vb + nodeOff[st] * CC,
          Mrel + ((size_t)l * 8 + e) * HH * DD * DD,
          albuf + alOff, segsum + nodeOff[dt] * HH, agg + nodeOff[dt] * CC);
      alOff += nw;
    }
    // gelu(agg) then output projection (WMMA)
    k_gelu<<<cdiv(NTOT * CC, 256), 256, 0, stream>>>(agg, NTOT * CC);
    for (int t = 0; t < 3; ++t) {
      const size_t wo = ((size_t)l * 4 + t) * CC * CC, bo = ((size_t)l * 4 + t) * CC;
      launch_gemm(agg + nodeOff[t] * CC, Wa + wo, ba + bo, xnew + nodeOff[t] * CC, Nt[t], CC, 0);
    }
    // layer 1: residual projections (agg space is free now)
    float* res0 = nullptr; float* res1 = nullptr;
    if (l == 1) {
      res0 = agg;
      res1 = agg + (size_t)Nt[0] * CC;
      launch_gemm(xcur, Wres_s, bres_s, res0, Nt[0], CC, 0);
      launch_gemm(xcur + nodeOff[1] * CC, Wres_o, bres_o, res1, Nt[1], CC, 0);
    }
    // combine: skip-gate + residual + leaky relu
    for (int t = 0; t < 3; ++t) {
      int mode = 0; const float* res = nullptr;
      if (l == 1) {
        if (t == 0) { mode = 1; res = res0; }
        else if (t == 1) { mode = 1; res = res1; }
        else mode = 2;  // start nodes: identity residual
      }
      const size_t n = (size_t)Nt[t] * CC;
      k_combine<<<cdiv(n, 256), 256, 0, stream>>>(xnew + nodeOff[t] * CC, xcur + nodeOff[t] * CC,
                                                  res, skip + (size_t)l * 4 + t, mode, n);
    }
  }

  // ---- output head: relu(xs0 @ Wo1 + bo1) @ Wo2 + bo2 ----
  float* hbuf = agg;  // NS*64 fits in agg
  launch_gemm(xcur, Wo1, bo1, hbuf, Nt[0], 64, 1);
  k_final<<<cdiv((size_t)Nt[0], 256), 256, 0, stream>>>(hbuf, Wo2, bo2, (float*)d_out, Nt[0]);
}
